// DSOTGraphBuilder_26749056320116
// MI455X (gfx1250) — matrix operand order repaired and checked by reading, not compiled, not run on hardware
//
#include <hip/hip_runtime.h>
#include <hip/hip_bf16.h>

// ---- problem constants (match reference setup_inputs) ----
#define B_ 16
#define N_ 1024
#define D_ 384
#define KNB 16
#define EPS_ 0.1f
#define SINK_ITERS 20
#define LAMBDA_POS 0.1f
#define ALPHA_ 0.2f
#define LOGR (-6.93147180559945309f)   // -ln(1024)

typedef __attribute__((ext_vector_type(16))) __bf16 v16bf;
typedef __attribute__((ext_vector_type(8)))  __bf16 v8bf;
typedef __attribute__((ext_vector_type(8)))  float  v8f;

// ---------------- K0: zero logu/logv/sums ----------------
__global__ void init_kernel(float* __restrict__ logu, float* __restrict__ logv,
                            float* __restrict__ sums) {
  int i = blockIdx.x * blockDim.x + threadIdx.x;
  if (i < B_ * N_) { logu[i] = 0.0f; logv[i] = 0.0f; }
  if (i < B_) sums[i] = 0.0f;
}

// ---------------- K1: L2-normalize rows, emit bf16 ----------------
__global__ __launch_bounds__(128)
void normalize_kernel(const float* __restrict__ x, __bf16* __restrict__ xn) {
  const int row = blockIdx.x;                 // [0, B*N)
  const float* xr = x + (size_t)row * D_;
  const int t = threadIdx.x;
  float v0 = xr[t], v1 = xr[t + 128], v2 = xr[t + 256];
  __shared__ float red[128];
  red[t] = v0 * v0 + v1 * v1 + v2 * v2;
  __syncthreads();
  for (int s = 64; s > 0; s >>= 1) {
    if (t < s) red[t] += red[t + s];
    __syncthreads();
  }
  const float sc = 1.0f / fmaxf(__builtin_sqrtf(red[0]), 1e-12f);
  __bf16* xo = xn + (size_t)row * D_;
  xo[t]       = (__bf16)(v0 * sc);
  xo[t + 128] = (__bf16)(v1 * sc);
  xo[t + 256] = (__bf16)(v2 * sc);
}

// ---------------- K2: WMMA GEMM -> cost tile + per-batch sum ----------------
// block = 256 threads = 8 waves; each wave computes one 16x16 tile of sim.
// grid = (N/128, N/16, B)
__global__ __launch_bounds__(256)
void cost_gemm_kernel(const __bf16* __restrict__ xn, const float* __restrict__ pos,
                      float* __restrict__ cost, float* __restrict__ sums) {
  const int b    = blockIdx.z;
  const int R    = blockIdx.y * 16;
  const int wave = threadIdx.x >> 5;
  const int lane = threadIdx.x & 31;
  const int C    = blockIdx.x * 128 + wave * 16;
  const int lh   = lane & 15;
  const int hi   = (lane >= 16) ? 1 : 0;

  const __bf16* xb   = xn + (size_t)b * N_ * D_;
  // A-frag: lane<16 holds row R+lane, K = k0..k0+15 contiguous;
  //         lane>=16 same row (lane-16), K shifted by +16 (matches B).
  const __bf16* arow = xb + (size_t)(R + lh) * D_ + (hi ? 16 : 0);
  // B-frag: lane<16 holds col C+lane with K=k0..k0+15 contiguous; lane>=16 K+=16.
  const __bf16* brow = xb + (size_t)(C + lh) * D_ + (hi ? 16 : 0);

  v8f acc = {};
#pragma unroll
  for (int k0 = 0; k0 < D_; k0 += 32) {
    v16bf a = *(const v16bf*)(arow + k0);
    v16bf bf = *(const v16bf*)(brow + k0);
    acc = __builtin_amdgcn_wmma_f32_16x16x32_bf16(
        false, a, false, bf, (short)0, acc, false, false);
  }

  const int j = C + lh;
  const float pjx = pos[2 * j], pjy = pos[2 * j + 1];
  float* cb = cost + (size_t)b * N_ * N_;
  float lsum = 0.0f;
#pragma unroll
  for (int v = 0; v < 8; ++v) {
    const int i = R + v + (hi ? 8 : 0);           // D-matrix row mapping
    const float dx = pos[2 * i] - pjx;
    const float dy = pos[2 * i + 1] - pjy;
    float d2 = dx * dx + dy * dy;
    if (d2 < 0.0f) d2 = 0.0f;
    const float c = 1.0f - acc[v] + LAMBDA_POS * d2;
    cb[(size_t)i * N_ + j] = c;
    lsum += c;
  }
  __shared__ float red[256];
  red[threadIdx.x] = lsum;
  __syncthreads();
  for (int s = 128; s > 0; s >>= 1) {
    if (threadIdx.x < s) red[threadIdx.x] += red[threadIdx.x + s];
    __syncthreads();
  }
  if (threadIdx.x == 0) atomicAdd(&sums[b], red[0]);
}

// ---------------- K3: cost -> logK = -cost / (eps*(mean+1e-8)) ----------------
__global__ void scale_kernel(float* __restrict__ cost, const float* __restrict__ sums) {
  size_t idx = (size_t)blockIdx.x * blockDim.x + threadIdx.x;
  const size_t total = (size_t)B_ * N_ * N_;
  const size_t stride = (size_t)gridDim.x * blockDim.x;
  for (; idx < total; idx += stride) {
    const int b = (int)(idx / ((size_t)N_ * N_));
    const float mean = sums[b] * (1.0f / (float)(N_ * N_));
    const float sc = -1.0f / (EPS_ * (mean + 1e-8f));
    cost[idx] *= sc;
  }
}

// ---------------- K4: logu[b,n] = LOGR - lse_m(logK[b,n,m] + logv[b,m]) --------
__global__ __launch_bounds__(256)
void row_lse_kernel(const float* __restrict__ logK, const float* __restrict__ logv,
                    float* __restrict__ logu) {
  const int n = blockIdx.x, b = blockIdx.y;
  const float* r  = logK + (size_t)b * N_ * N_ + (size_t)n * N_;
  const float* lv = logv + (size_t)b * N_;
  float mx = -__builtin_inff(), s = 0.0f;
  for (int m = threadIdx.x; m < N_; m += 256) {
    const float v = r[m] + lv[m];
    if (v > mx) { s = s * __expf(mx - v) + 1.0f; mx = v; }
    else         { s += __expf(v - mx); }
  }
  __shared__ float smx[256], ss[256];
  smx[threadIdx.x] = mx; ss[threadIdx.x] = s;
  __syncthreads();
  for (int st = 128; st > 0; st >>= 1) {
    if (threadIdx.x < st) {
      const float m1 = smx[threadIdx.x], s1 = ss[threadIdx.x];
      const float m2 = smx[threadIdx.x + st], s2 = ss[threadIdx.x + st];
      const float M = fmaxf(m1, m2);
      const float S = (M == -__builtin_inff()) ? 0.0f
                    : s1 * __expf(m1 - M) + s2 * __expf(m2 - M);
      smx[threadIdx.x] = M; ss[threadIdx.x] = S;
    }
    __syncthreads();
  }
  if (threadIdx.x == 0)
    logu[(size_t)b * N_ + n] = LOGR - (smx[0] + __logf(ss[0]));
}

// ---------------- K5: logv[b,m] = LOGR - lse_n(logK[b,n,m] + logu[b,n]) --------
// one thread per column m -> fully coalesced row-major streaming
__global__ __launch_bounds__(256)
void col_lse_kernel(const float* __restrict__ logK, const float* __restrict__ logu,
                    float* __restrict__ logv) {
  const int b = blockIdx.y;
  const int m = blockIdx.x * 256 + threadIdx.x;
  __shared__ float lu[N_];
  for (int i = threadIdx.x; i < N_; i += 256) lu[i] = logu[(size_t)b * N_ + i];
  __syncthreads();
  const float* base = logK + (size_t)b * N_ * N_ + m;
  float mx = -__builtin_inff(), s = 0.0f;
  for (int n = 0; n < N_; ++n) {
    const float v = base[(size_t)n * N_] + lu[n];
    if (v > mx) { s = s * __expf(mx - v) + 1.0f; mx = v; }
    else         { s += __expf(v - mx); }
  }
  logv[(size_t)b * N_ + m] = LOGR - (mx + __logf(s));
}

// ---------------- K6: A = 0.5(P+P^T)+alpha*I, top-16, emit edges ----------------
__global__ __launch_bounds__(256)
void topk_kernel(const float* __restrict__ logK, const float* __restrict__ logu,
                 const float* __restrict__ logv, float* __restrict__ out) {
  const int n = blockIdx.x, b = blockIdx.y;
  __shared__ float arow[N_];
  __shared__ float rv[256];
  __shared__ int   ri[256];
  __shared__ float topv[KNB];
  __shared__ int   topi[KNB];

  const float* lk = logK + (size_t)b * N_ * N_;
  const float* lu = logu + (size_t)b * N_;
  const float* lv = logv + (size_t)b * N_;
  const float lun = lu[n], lvn = lv[n];
  for (int m = threadIdx.x; m < N_; m += 256) {
    const float p  = __expf(lun + lk[(size_t)n * N_ + m] + lv[m]);
    const float pt = __expf(lu[m] + lk[(size_t)m * N_ + n] + lvn);
    float a = 0.5f * (p + pt);
    if (m == n) a += ALPHA_;
    arow[m] = a;
  }
  __syncthreads();

  for (int t = 0; t < KNB; ++t) {
    float bm = -__builtin_inff(); int bi = N_;
    for (int m = threadIdx.x; m < N_; m += 256) {
      const float v = arow[m];
      if (v > bm || (v == bm && m < bi)) { bm = v; bi = m; }
    }
    rv[threadIdx.x] = bm; ri[threadIdx.x] = bi;
    __syncthreads();
    for (int st = 128; st > 0; st >>= 1) {
      if (threadIdx.x < st) {
        const float v1 = rv[threadIdx.x], v2 = rv[threadIdx.x + st];
        const int   i1 = ri[threadIdx.x], i2 = ri[threadIdx.x + st];
        if (v2 > v1 || (v2 == v1 && i2 < i1)) { rv[threadIdx.x] = v2; ri[threadIdx.x] = i2; }
      }
      __syncthreads();
    }
    if (threadIdx.x == 0) { topv[t] = rv[0]; topi[t] = ri[0]; arow[ri[0]] = -__builtin_inff(); }
    __syncthreads();
  }

  if (threadIdx.x == 0) {
    float sm = 0.0f;
    for (int t = 0; t < KNB; ++t) sm += topv[t];
    rv[0] = sm;
  }
  __syncthreads();
  const float inv = 1.0f / (rv[0] + 1e-12f);
  const size_t BNK = (size_t)B_ * N_ * KNB;
  if (threadIdx.x < KNB) {
    const int j = threadIdx.x;
    const size_t e = ((size_t)b * N_ + n) * KNB + j;
    out[e]           = (float)(b * N_ + n);            // edge_index[0] (src)
    out[BNK + e]     = (float)(topi[j] + b * N_);      // edge_index[1] (dst)
    out[2 * BNK + e] = topv[j] * inv;                  // edge_weight
  }
  if (threadIdx.x == 0)
    out[3 * BNK + (size_t)b * N_ + n] = (float)b;      // batch
}

// ---------------- host launcher ----------------
extern "C" void kernel_launch(void* const* d_in, const int* in_sizes, int n_in,
                              void* d_out, int out_size, void* d_ws, size_t ws_size,
                              hipStream_t stream) {
  (void)in_sizes; (void)n_in; (void)out_size; (void)ws_size;
  const float* x   = (const float*)d_in[0];   // (B, N, D) f32
  const float* pos = (const float*)d_in[1];   // (N, 2)    f32
  float* out = (float*)d_out;

  // workspace layout (bytes): cost/logK fp32 64MB | xn bf16 12.6MB | logu | logv | sums
  char* ws = (char*)d_ws;
  float*  cost = (float*)ws;                                   // B*N*N
  __bf16* xn   = (__bf16*)(ws + (size_t)67108864);             // B*N*D
  float*  logu = (float*)(ws + (size_t)79691776);              // B*N
  float*  logv = (float*)(ws + (size_t)79757312);              // B*N
  float*  sums = (float*)(ws + (size_t)79822848);              // B

  init_kernel<<<(B_ * N_) / 256, 256, 0, stream>>>(logu, logv, sums);
  normalize_kernel<<<B_ * N_, 128, 0, stream>>>(x, xn);

  dim3 g2(N_ / 128, N_ / 16, B_);
  cost_gemm_kernel<<<g2, 256, 0, stream>>>(xn, pos, cost, sums);

  scale_kernel<<<4096, 256, 0, stream>>>(cost, sums);

  for (int it = 0; it < SINK_ITERS; ++it) {
    row_lse_kernel<<<dim3(N_, B_), 256, 0, stream>>>(cost, logv, logu);
    col_lse_kernel<<<dim3(N_ / 256, B_), 256, 0, stream>>>(cost, logu, logv);
  }

  topk_kernel<<<dim3(N_, B_), 256, 0, stream>>>(cost, logu, logv, out);
}
